// ColumnGraphNet_39419209842758
// MI455X (gfx1250) — matrix operand order repaired and checked by reading, not compiled, not run on hardware
//
#include <hip/hip_runtime.h>
#include <hip/hip_bf16.h>

typedef __attribute__((ext_vector_type(16))) _Float16 v16h;
typedef __attribute__((ext_vector_type(8)))  _Float16 v8h;
typedef __attribute__((ext_vector_type(4)))  _Float16 v4h;
typedef __attribute__((ext_vector_type(8)))  float    v8f;

#define NCELL 172032   // 4096*42
#define NCOL  28672    // 4096*7
#define NB    4096
#define ROWS  6
#define BW    7

// ---- params flat-buffer offsets (jax tree order: dict keys sorted) ----
#define L_BASE   256
#define L_STRIDE 39438
#define O_AW_CELL 0
#define O_AW_COL  4096
#define O_A_REL   8192      // [3][4][16][16]
#define O_AB_CELL 11264
#define O_AB_COL  11328
#define O_KW_CELL 11392
#define O_KW_COL  15488
#define O_KB_CELL 19584
#define O_KB_COL  19648
#define O_M_REL   19712
#define O_P_REL   22784     // [3][4]
#define O_QW_CELL 22796
#define O_QW_COL  26892
#define O_QB_CELL 30988
#define O_QB_COL  31052
#define O_SKIP_CELL 31116
#define O_SKIP_COL  31117
#define O_VW_CELL 31118
#define O_VW_COL  35214
#define O_VB_CELL 39310
#define O_VB_COL  39374

// ---------------------------------------------------------------------------
// prep: fold block-diagonal relation transforms into projection weights,
// convert to f16, and store each 64x64 matrix PRE-SWIZZLED into the WMMA
// B-fragment register image: [frag f=kk*4+nb][lane][e] with
//   lane = 16*half + n,  K = kk*32 + half*16 + e,  col = nb*16 + n.
// Each lane's 16 halves are contiguous -> one v16h global load per fragment.
// slot map/layer: 0 q_cell, 1 k0, 2 v0, 3 k1, 4 v1, 5 q_col, 6 k2, 7 v2,
//                 8 a_cell, 9 a_col; slot 20 pfc, 21 vfc.
// ---------------------------------------------------------------------------
__global__ __launch_bounds__(256) void prep_kernel(const float* __restrict__ P,
                                                   _Float16* __restrict__ W16,
                                                   float* __restrict__ B32) {
  int m = blockIdx.x;
  const float* Wsrc = P; const float* bsrc = P; const float* A = nullptr;
  if (m < 20) {
    int l = m / 10, s = m % 10;
    const float* L = P + L_BASE + (size_t)l * L_STRIDE;
    switch (s) {
      case 0: Wsrc = L + O_QW_CELL; bsrc = L + O_QB_CELL; break;
      case 1: Wsrc = L + O_KW_CELL; bsrc = L + O_KB_CELL; A = L + O_A_REL;        break;
      case 2: Wsrc = L + O_VW_CELL; bsrc = L + O_VB_CELL; A = L + O_M_REL;        break;
      case 3: Wsrc = L + O_KW_CELL; bsrc = L + O_KB_CELL; A = L + O_A_REL + 1024; break;
      case 4: Wsrc = L + O_VW_CELL; bsrc = L + O_VB_CELL; A = L + O_M_REL + 1024; break;
      case 5: Wsrc = L + O_QW_COL;  bsrc = L + O_QB_COL;  break;
      case 6: Wsrc = L + O_KW_COL;  bsrc = L + O_KB_COL;  A = L + O_A_REL + 2048; break;
      case 7: Wsrc = L + O_VW_COL;  bsrc = L + O_VB_COL;  A = L + O_M_REL + 2048; break;
      case 8: Wsrc = L + O_AW_CELL; bsrc = L + O_AB_CELL; break;
      default: Wsrc = L + O_AW_COL; bsrc = L + O_AB_COL;  break;
    }
  } else if (m == 20) { Wsrc = P + 79260; bsrc = P + 83356; }
  else               { Wsrc = P + 83485; bsrc = P + 87581; }

  _Float16* Wout = W16 + (size_t)m * 4096;
  float*    Bout = B32 + (size_t)m * 64;
  for (int idx = threadIdx.x; idx < 4096; idx += 256) {
    int f    = idx >> 9;          // fragment 0..7
    int lane = (idx >> 4) & 31;
    int e    = idx & 15;
    int halfId = lane >> 4, n = lane & 15;
    int kk = f >> 2, nb = f & 3;
    int k   = kk * 32 + halfId * 16 + e;
    int col = nb * 16 + n;
    float val;
    if (!A) {
      val = Wsrc[k * 64 + col];
    } else {
      int h = col >> 4, e2 = col & 15;
      float s = 0.f;
      #pragma unroll
      for (int d = 0; d < 16; ++d) s += Wsrc[k * 64 + h * 16 + d] * A[h * 256 + d * 16 + e2];
      val = s;
    }
    Wout[idx] = (_Float16)val;
  }
  if (threadIdx.x < 64) {
    int col = threadIdx.x;
    if (!A) {
      Bout[col] = bsrc[col];
    } else {
      int h = col >> 4, e = col & 15;
      float s = 0.f;
      #pragma unroll
      for (int d = 0; d < 16; ++d) s += bsrc[h * 16 + d] * A[h * 256 + d * 16 + e];
      Bout[col] = s;
    }
  }
}

// ---------------------------------------------------------------------------
// input projections
// ---------------------------------------------------------------------------
__global__ __launch_bounds__(256) void inproj_cell_kernel(const float* __restrict__ cx,
                                                          const float* __restrict__ P,
                                                          float* __restrict__ xc) {
  int t = blockIdx.x * 256 + threadIdx.x;           // NCELL*64
  int i = t >> 6, ch = t & 63;
  const float* W = P;            // cin_W [3][64]
  const float* b = P + 192;      // cin_b
  xc[t] = cx[i * 3 + 0] * W[ch] + cx[i * 3 + 1] * W[64 + ch] + cx[i * 3 + 2] * W[128 + ch] + b[ch];
}

__global__ __launch_bounds__(256) void inproj_col_kernel(const float* __restrict__ lx,
                                                         const float* __restrict__ P,
                                                         float* __restrict__ xl) {
  int t = blockIdx.x * 256 + threadIdx.x;           // NCOL*64
  int i = t >> 6, ch = t & 63;
  xl[t] = lx[i] * P[79132 + ch] + P[79196 + ch];
}

__device__ __forceinline__ float gelu_f(float x) {
  return 0.5f * x * (1.f + erff(x * 0.70710678118f));
}

// ---------------------------------------------------------------------------
// WMMA GEMM: Y = act(X[rows x 64] @ W[64x64] + bias). 128 rows/block, 8 waves,
// each wave owns a private 16-row tile: stages X (f32->f16) into its own LDS
// region, loads B fragments as v16h from pre-swizzled global weights, runs
// 2 k-steps x 4 N-tiles of v_wmma_f32_16x16x32_f16, stages C through LDS and
// writes back 128-bit coalesced. No cross-wave sharing -> no barriers.
// MODE 0: f16 out (z-indexed);  MODE 1: gelu-in + skip blend f32;  MODE 2: relu f32
// ---------------------------------------------------------------------------
template <int MODE>
__global__ __launch_bounds__(256) void gemm64_kernel(const float* __restrict__ X,
                                                     const _Float16* __restrict__ Wb,
                                                     const float* __restrict__ Bb,
                                                     void* __restrict__ Yb,
                                                     long rowsPerMat,
                                                     const float* __restrict__ Xold,
                                                     const float* __restrict__ skipP) {
  __shared__ _Float16 sX[128 * 64];   // 16KB: staged A operand (f16)
  __shared__ float    sO[128 * 64];   // 32KB: C staging (f32, or f16 view in MODE 0)

  const int tid  = threadIdx.x;
  const int lane = tid & 31, wave = tid >> 5;
  const int halfId = lane >> 4, mn = lane & 15;
  const int rBase  = wave * 16;
  const int z = blockIdx.y;
  const size_t rowBlock = (size_t)blockIdx.x * 128;

  // ---- stage this wave's 16 rows of X: f32 -> f16 (optional gelu), b64 LDS stores
  {
    const float4* src = (const float4*)(X + (rowBlock + rBase) * 64);
    _Float16* dstBase = sX + rBase * 64;
    #pragma unroll
    for (int j = 0; j < 8; ++j) {
      int idx = j * 32 + lane;               // 256 float4 per wave tile
      float4 v = src[idx];
      if (MODE == 1) {
        v.x = gelu_f(v.x); v.y = gelu_f(v.y); v.z = gelu_f(v.z); v.w = gelu_f(v.w);
      }
      v4h hv;
      hv[0] = (_Float16)v.x; hv[1] = (_Float16)v.y;
      hv[2] = (_Float16)v.z; hv[3] = (_Float16)v.w;
      *(v4h*)(dstBase + idx * 4) = hv;
    }
  }

  const v16h* Wf = (const v16h*)(Wb + (size_t)z * 4096);   // 8 fragments x 32 lanes

  v8f acc[4];
  #pragma unroll
  for (int nb = 0; nb < 4; ++nb) acc[nb] = (v8f){0.f, 0.f, 0.f, 0.f, 0.f, 0.f, 0.f, 0.f};

  #pragma unroll
  for (int kk = 0; kk < 2; ++kk) {
    // A fragment = one contiguous 16-half run: K = 16*half + {0..15}
    const _Float16* xrow = sX + (rBase + mn) * 64 + kk * 32 + halfId * 16;
    v8h lo = *(const v8h*)(xrow);
    v8h hi = *(const v8h*)(xrow + 8);
    v16h a = __builtin_shufflevector(lo, hi, 0, 1, 2, 3, 4, 5, 6, 7,
                                             8, 9, 10, 11, 12, 13, 14, 15);
    #pragma unroll
    for (int nb = 0; nb < 4; ++nb) {
      v16h b = Wf[(kk * 4 + nb) * 32 + lane];              // pre-swizzled register image
      acc[nb] = __builtin_amdgcn_wmma_f32_16x16x32_f16(false, a, false, b, (short)0,
                                                       acc[nb], false, false);
    }
  }

  const float* bias = Bb + (size_t)z * 64;
  const int rrow  = lane >> 1;            // readback: 2 lanes per row
  const int chalf = (lane & 1) * 32;

  if (MODE == 0) {
    // stage C as f16 into this wave's region of sO, then coalesced b128 stores
    _Float16* st = ((_Float16*)sO) + rBase * 64;
    #pragma unroll
    for (int nb = 0; nb < 4; ++nb) {
      float bv = bias[nb * 16 + mn];
      #pragma unroll
      for (int r = 0; r < 8; ++r)
        st[(r + 8 * halfId) * 64 + nb * 16 + mn] = (_Float16)(acc[nb][r] + bv);
    }
    const v8h* rb = (const v8h*)(((_Float16*)sO) + (rBase + rrow) * 64 + chalf);
    _Float16* Y = (_Float16*)Yb +
                  ((size_t)z * (size_t)rowsPerMat + rowBlock + rBase + rrow) * 64 + chalf;
    #pragma unroll
    for (int j = 0; j < 4; ++j) ((v8h*)Y)[j] = rb[j];
  } else {
    float* st = sO + rBase * 64;
    #pragma unroll
    for (int nb = 0; nb < 4; ++nb) {
      float bv = bias[nb * 16 + mn];
      #pragma unroll
      for (int r = 0; r < 8; ++r)
        st[(r + 8 * halfId) * 64 + nb * 16 + mn] = acc[nb][r] + bv;
    }
    float s = 0.f, oneMs = 0.f;
    if (MODE == 1) {
      float sk = *skipP;
      s = 1.f / (1.f + expf(-sk));
      oneMs = 1.f - s;
    }
    const float4* rb = (const float4*)(sO + (rBase + rrow) * 64 + chalf);
    size_t gidx = (rowBlock + rBase + rrow) * 64 + chalf;
    float4* Y = (float4*)((float*)Yb + gidx);
    const float4* xo = (const float4*)(Xold + gidx);   // valid only in MODE 1
    #pragma unroll
    for (int j = 0; j < 8; ++j) {
      float4 v = rb[j];
      if (MODE == 1) {
        float4 o = xo[j];
        v.x = s * v.x + oneMs * o.x; v.y = s * v.y + oneMs * o.y;
        v.z = s * v.z + oneMs * o.z; v.w = s * v.w + oneMs * o.w;
      } else {
        v.x = v.x > 0.f ? v.x : 0.f; v.y = v.y > 0.f ? v.y : 0.f;
        v.z = v.z > 0.f ? v.z : 0.f; v.w = v.w > 0.f ? v.w : 0.f;
      }
      Y[j] = v;
    }
  }
}

// ---------------------------------------------------------------------------
// attention: destination = cell. Up to 4 adjacency edges (rel 0) + rev edge
// from its column node (rel 2), joint softmax. One thread per (cell, head).
// ---------------------------------------------------------------------------
__global__ __launch_bounds__(256) void attn_cell_kernel(const _Float16* __restrict__ Qc,
                                                        const _Float16* __restrict__ K0,
                                                        const _Float16* __restrict__ V0,
                                                        const _Float16* __restrict__ K2,
                                                        const _Float16* __restrict__ V2,
                                                        const float* __restrict__ prel,
                                                        float* __restrict__ agg) {
  int t = blockIdx.x * 256 + threadIdx.x;          // NCELL*4
  int h = t & 3, cell = t >> 2;
  int b = cell / 42, rc = cell - b * 42;
  int r = rc / 7, c = rc - r * 7;

  v16h qv = *(const v16h*)(Qc + (size_t)cell * 64 + h * 16);
  float q[16];
  #pragma unroll
  for (int j = 0; j < 16; ++j) q[j] = (float)qv[j];

  const float scale = 0.25f;  // 1/sqrt(16)
  float p0 = prel[0 * 4 + h] * scale;
  float p2 = prel[2 * 4 + h] * scale;

  const _Float16* kp[5]; const _Float16* vp[5]; float pw[5];
  int n = 0;
  if (c > 0)        { size_t s = (size_t)(cell - 1) * 64 + h * 16; kp[n] = K0 + s; vp[n] = V0 + s; pw[n] = p0; ++n; }
  if (c < BW - 1)   { size_t s = (size_t)(cell + 1) * 64 + h * 16; kp[n] = K0 + s; vp[n] = V0 + s; pw[n] = p0; ++n; }
  if (r > 0)        { size_t s = (size_t)(cell - 7) * 64 + h * 16; kp[n] = K0 + s; vp[n] = V0 + s; pw[n] = p0; ++n; }
  if (r < ROWS - 1) { size_t s = (size_t)(cell + 7) * 64 + h * 16; kp[n] = K0 + s; vp[n] = V0 + s; pw[n] = p0; ++n; }
  { size_t s = (size_t)(b * 7 + c) * 64 + h * 16;                 kp[n] = K2 + s; vp[n] = V2 + s; pw[n] = p2; ++n; }

  float logit[5], mx = -1e30f;
  for (int i = 0; i < n; ++i) {
    v16h kv = *(const v16h*)kp[i];
    float d = 0.f;
    #pragma unroll
    for (int j = 0; j < 16; ++j) d += q[j] * (float)kv[j];
    logit[i] = d * pw[i];
    mx = fmaxf(mx, logit[i]);
  }
  float denom = 0.f, accv[16];
  #pragma unroll
  for (int j = 0; j < 16; ++j) accv[j] = 0.f;
  for (int i = 0; i < n; ++i) {
    float e = expf(logit[i] - mx);
    denom += e;
    v16h vv = *(const v16h*)vp[i];
    #pragma unroll
    for (int j = 0; j < 16; ++j) accv[j] += e * (float)vv[j];
  }
  float inv = 1.f / (denom + 1e-16f);
  float* dst = agg + (size_t)cell * 64 + h * 16;
  #pragma unroll
  for (int j = 0; j < 16; ++j) dst[j] = accv[j] * inv;
}

// destination = column node: 6 incoming edges from cells of that column (rel 1)
__global__ __launch_bounds__(256) void attn_col_kernel(const _Float16* __restrict__ Ql,
                                                       const _Float16* __restrict__ K1,
                                                       const _Float16* __restrict__ V1,
                                                       const float* __restrict__ prel,
                                                       float* __restrict__ agg) {
  int t = blockIdx.x * 256 + threadIdx.x;          // NCOL*4
  int h = t & 3, col = t >> 2;
  int b = col / 7, c = col - b * 7;

  v16h qv = *(const v16h*)(Ql + (size_t)col * 64 + h * 16);
  float q[16];
  #pragma unroll
  for (int j = 0; j < 16; ++j) q[j] = (float)qv[j];
  float p1 = prel[1 * 4 + h] * 0.25f;

  float logit[ROWS], mx = -1e30f;
  #pragma unroll
  for (int i = 0; i < ROWS; ++i) {
    size_t s = ((size_t)(b * 42 + i * 7 + c)) * 64 + h * 16;
    v16h kv = *(const v16h*)(K1 + s);
    float d = 0.f;
    #pragma unroll
    for (int j = 0; j < 16; ++j) d += q[j] * (float)kv[j];
    logit[i] = d * p1;
    mx = fmaxf(mx, logit[i]);
  }
  float denom = 0.f, accv[16];
  #pragma unroll
  for (int j = 0; j < 16; ++j) accv[j] = 0.f;
  #pragma unroll
  for (int i = 0; i < ROWS; ++i) {
    float e = expf(logit[i] - mx);
    denom += e;
    size_t s = ((size_t)(b * 42 + i * 7 + c)) * 64 + h * 16;
    v16h vv = *(const v16h*)(V1 + s);
    #pragma unroll
    for (int j = 0; j < 16; ++j) accv[j] += e * (float)vv[j];
  }
  float inv = 1.f / (denom + 1e-16f);
  float* dst = agg + (size_t)col * 64 + h * 16;
  #pragma unroll
  for (int j = 0; j < 16; ++j) dst[j] = accv[j] * inv;
}

// ---------------------------------------------------------------------------
// heads
// ---------------------------------------------------------------------------
__global__ __launch_bounds__(256) void pool_kernel(const float* __restrict__ xc,
                                                   float* __restrict__ g) {
  int t = blockIdx.x * 256 + threadIdx.x;          // NB*64
  int b = t >> 6, ch = t & 63;
  size_t base = (size_t)b * 42 * 64 + ch;
  float s = 0.f;
  #pragma unroll
  for (int i = 0; i < 42; ++i) s += xc[base + (size_t)i * 64];
  g[t] = s;
}

__global__ __launch_bounds__(256) void policy_out_kernel(const float* __restrict__ ph,
                                                         const float* __restrict__ P,
                                                         float* __restrict__ out) {
  int i = blockIdx.x * 256 + threadIdx.x;          // NCOL
  const float* row = ph + (size_t)i * 64;
  const float* w = P + 83420;                       // ph_W
  float s = P[83484];                               // ph_b
  #pragma unroll
  for (int j = 0; j < 64; ++j) s += row[j] * w[j];
  out[i] = s;                                       // [B,7] flat == col-node id
}

__global__ __launch_bounds__(256) void value_out_kernel(const float* __restrict__ vbuf,
                                                        const float* __restrict__ P,
                                                        float* __restrict__ out) {
  int b = blockIdx.x * 256 + threadIdx.x;          // NB
  const float* row = vbuf + (size_t)b * 64;
  const float* w = P + 87645;                       // vh_W
  float s = P[87709];                               // vh_b
  #pragma unroll
  for (int j = 0; j < 64; ++j) s += row[j] * w[j];
  out[NCOL + b] = tanhf(s);
}

// ---------------------------------------------------------------------------
extern "C" void kernel_launch(void* const* d_in, const int* in_sizes, int n_in,
                              void* d_out, int out_size, void* d_ws, size_t ws_size,
                              hipStream_t stream) {
  const float* cell_x = (const float*)d_in[0];
  const float* col_x  = (const float*)d_in[1];
  const float* P      = (const float*)d_in[2];   // flattened params pytree
  float* out = (float*)d_out;                    // [NCOL policy | NB value]

  char* w = (char*)d_ws;
  auto carve = [&](size_t bytes) -> void* {
    void* p = (void*)w;
    w += (bytes + 255) & ~(size_t)255;
    return p;
  };
  _Float16* W16  = (_Float16*)carve(22 * 4096 * sizeof(_Float16));
  float*    B32  = (float*)   carve(22 * 64 * sizeof(float));
  float*    xc   = (float*)   carve((size_t)NCELL * 64 * sizeof(float));
  float*    aggc = (float*)   carve((size_t)NCELL * 64 * sizeof(float));
  float*    xl   = (float*)   carve((size_t)NCOL * 64 * sizeof(float));
  float*    aggl = (float*)   carve((size_t)NCOL * 64 * sizeof(float));
  _Float16* KQVc = (_Float16*)carve((size_t)5 * NCELL * 64 * sizeof(_Float16));
  _Float16* KQVl = (_Float16*)carve((size_t)3 * NCOL * 64 * sizeof(_Float16));
  float*    ph   = (float*)   carve((size_t)NCOL * 64 * sizeof(float));
  float*    g    = (float*)   carve((size_t)NB * 64 * sizeof(float));
  float*    vbuf = (float*)   carve((size_t)NB * 64 * sizeof(float));

  prep_kernel<<<22, 256, 0, stream>>>(P, W16, B32);

  inproj_cell_kernel<<<(NCELL * 64) / 256, 256, 0, stream>>>(cell_x, P, xc);
  inproj_col_kernel<<<(NCOL * 64) / 256, 256, 0, stream>>>(col_x, P, xl);

  for (int l = 0; l < 2; ++l) {
    const float* L      = P + L_BASE + (size_t)l * L_STRIDE;
    const float* prel   = L + O_P_REL;
    const float* skip_c = L + O_SKIP_CELL;
    const float* skip_l = L + O_SKIP_COL;
    const _Float16* Wc = W16 + (size_t)(l * 10 + 0) * 4096;  // slots 0..4: Qc,K0,V0,K1,V1
    const float*    Bc = B32 + (size_t)(l * 10 + 0) * 64;
    const _Float16* Wl = W16 + (size_t)(l * 10 + 5) * 4096;  // slots 5..7: Ql,K2,V2
    const float*    Bl = B32 + (size_t)(l * 10 + 5) * 64;

    gemm64_kernel<0><<<dim3(NCELL / 128, 5), 256, 0, stream>>>(
        xc, Wc, Bc, (void*)KQVc, (long)NCELL, nullptr, nullptr);
    gemm64_kernel<0><<<dim3(NCOL / 128, 3), 256, 0, stream>>>(
        xl, Wl, Bl, (void*)KQVl, (long)NCOL, nullptr, nullptr);

    const _Float16* Qc = KQVc + (size_t)0 * NCELL * 64;
    const _Float16* K0 = KQVc + (size_t)1 * NCELL * 64;
    const _Float16* V0 = KQVc + (size_t)2 * NCELL * 64;
    const _Float16* K1 = KQVc + (size_t)3 * NCELL * 64;
    const _Float16* V1 = KQVc + (size_t)4 * NCELL * 64;
    const _Float16* Ql = KQVl + (size_t)0 * NCOL * 64;
    const _Float16* K2 = KQVl + (size_t)1 * NCOL * 64;
    const _Float16* V2 = KQVl + (size_t)2 * NCOL * 64;

    attn_cell_kernel<<<(NCELL * 4) / 256, 256, 0, stream>>>(Qc, K0, V0, K2, V2, prel, aggc);
    attn_col_kernel<<<(NCOL * 4) / 256, 256, 0, stream>>>(Ql, K1, V1, prel, aggl);

    gemm64_kernel<1><<<dim3(NCELL / 128, 1), 256, 0, stream>>>(
        aggc, W16 + (size_t)(l * 10 + 8) * 4096, B32 + (size_t)(l * 10 + 8) * 64,
        (void*)xc, (long)NCELL, xc, skip_c);
    gemm64_kernel<1><<<dim3(NCOL / 128, 1), 256, 0, stream>>>(
        aggl, W16 + (size_t)(l * 10 + 9) * 4096, B32 + (size_t)(l * 10 + 9) * 64,
        (void*)xl, (long)NCOL, xl, skip_l);
  }

  gemm64_kernel<2><<<dim3(NCOL / 128, 1), 256, 0, stream>>>(
      xl, W16 + (size_t)20 * 4096, B32 + (size_t)20 * 64, (void*)ph, (long)NCOL,
      nullptr, nullptr);
  policy_out_kernel<<<NCOL / 256, 256, 0, stream>>>(ph, P, out);

  pool_kernel<<<(NB * 64) / 256, 256, 0, stream>>>(xc, g);
  gemm64_kernel<2><<<dim3(NB / 128, 1), 256, 0, stream>>>(
      g, W16 + (size_t)21 * 4096, B32 + (size_t)21 * 64, (void*)vbuf, (long)NB,
      nullptr, nullptr);
  value_out_kernel<<<NB / 256, 256, 0, stream>>>(vbuf, P, out);
}
